// BiRGAT_1056561955275
// MI455X (gfx1250) — matrix operand order repaired and checked by reading, not compiled, not run on hardware
//
#include <hip/hip_runtime.h>
#include <hip/hip_bf16.h>
#include <stdint.h>

// ---------------------------------------------------------------------------
// BiRGAT forward for gfx1250 (MI455X): bf16-WMMA GEMMs (pre-converted bf16
// operands, 256x64 block tiles, LDS-staged weights, preloaded B fragments)
// + coalesced edge softmax/scatter with f32 atomics + fused epilogues.
// ---------------------------------------------------------------------------

#define KD  256   // inner dim for every dense projection in this model
#define HC  256   // H*C
#define CH  64    // per-head channels
#define NH  4     // heads
#define KDP 264   // padded LDS row stride (ushorts): 528B keeps 16B alignment,
                  // lane bank stride 132 dwords -> banks l16*4 (conflict-free b128)

typedef __attribute__((ext_vector_type(16))) __bf16        v16bf;
typedef __attribute__((ext_vector_type(8)))  float         v8f;
typedef __attribute__((ext_vector_type(4)))  float         f32x4;
typedef __attribute__((ext_vector_type(4)))  unsigned int  u32x4;

union BF16x16 {
    v16bf          v;
    unsigned short u[16];
    u32x4          q[2];
};

union BF16x8 {
    u32x4          q;
    __bf16         h[8];
    unsigned short u[8];
};

// order-preserving float<->int map for integer atomicMax on f32 scores
__device__ inline int   encf(float f) { int i = __float_as_int(f); return i >= 0 ? i : (i ^ 0x7FFFFFFF); }
__device__ inline float decf(int i)   { return __int_as_float(i >= 0 ? i : (i ^ 0x7FFFFFFF)); }

// ---------------------------------------------------------------------------
// f32 -> bf16 bulk convert (8 elems/thread, hardware cvt). n % 8 == 0.
// ---------------------------------------------------------------------------
__global__ void cvt_f32_bf16(const float* __restrict__ in,
                             unsigned short* __restrict__ out, int n)
{
    int i = (blockIdx.x * blockDim.x + threadIdx.x) * 8;
    if (i >= n) return;
    f32x4 f0 = *reinterpret_cast<const f32x4*>(in + i);
    f32x4 f1 = *reinterpret_cast<const f32x4*>(in + i + 4);
    BF16x8 o;
    o.h[0] = (__bf16)f0.x; o.h[1] = (__bf16)f0.y;
    o.h[2] = (__bf16)f0.z; o.h[3] = (__bf16)f0.w;
    o.h[4] = (__bf16)f1.x; o.h[5] = (__bf16)f1.y;
    o.h[6] = (__bf16)f1.z; o.h[7] = (__bf16)f1.w;
    *reinterpret_cast<u32x4*>(out + i) = o.q;
}

// ---------------------------------------------------------------------------
// GEMM: Y[nrows,M] = X[nrows,KD] @ W[KD,M] + bias   (bf16 in, f32 accum)
// block = 256 threads = 8 waves; block tile = 256 rows x 64 cols.
// Each wave: 2 A row-fragments (rows w*16 and w*16+128) x 4 B col-tiles
// = 8 accumulators, 64 WMMAs/wave. B fragments preloaded per k-step so the
// 8 ds_load_b128 clause together; each B feeds 2 WMMAs.
// ---------------------------------------------------------------------------
__global__ __launch_bounds__(256)
void gemm_bf16_wmma(const unsigned short* __restrict__ X,
                    const unsigned short* __restrict__ Wb,
                    const float* __restrict__ bias, float* __restrict__ Y,
                    int nrows, int M)
{
    __shared__ unsigned short wt[64 * KDP];          // [col][k] bf16, padded
    const int tid  = threadIdx.x;
    const int col0 = blockIdx.x << 6;

    for (int i = tid; i < 64 * KD; i += 256) {       // coalesced: consecutive cols
        int k = i >> 6;
        int c = i & 63;
        wt[c * KDP + k] = Wb[(size_t)k * M + col0 + c];
    }
    __syncthreads();

    const int lane = tid & 31;
    const int half = lane >> 4;
    const int l16  = lane & 15;
    const int rbase = (blockIdx.y << 8) + ((tid >> 5) << 4);  // wave rows: rbase, rbase+128

    int ar0 = rbase + l16;        if (ar0 >= nrows) ar0 = nrows - 1;  // clamp: EXEC stays full
    int ar1 = rbase + 128 + l16;  if (ar1 >= nrows) ar1 = nrows - 1;
    const unsigned short* xr0 = X + (size_t)ar0 * KD;
    const unsigned short* xr1 = X + (size_t)ar1 * KD;

    v8f acc0[4] = {};
    v8f acc1[4] = {};
#pragma unroll
    for (int k0 = 0; k0 < KD; k0 += 32) {
        // A 16x32 bf16 layout: elems 0-7 -> K = k0+8*half+{0..7}; 8-15 -> +16
        BF16x16 a0, a1;
        a0.q[0] = *reinterpret_cast<const u32x4*>(xr0 + k0 + half * 8);
        a0.q[1] = *reinterpret_cast<const u32x4*>(xr0 + k0 + 16 + half * 8);
        a1.q[0] = *reinterpret_cast<const u32x4*>(xr1 + k0 + half * 8);
        a1.q[1] = *reinterpret_cast<const u32x4*>(xr1 + k0 + 16 + half * 8);
        // B 32x16 bf16 layout: lane col = l16, elems 0-7 -> K = k0+8*half+{0..7}; 8-15 -> +16
        BF16x16 b[4];
#pragma unroll
        for (int ct = 0; ct < 4; ++ct) {
            const unsigned short* wl = wt + (ct * 16 + l16) * KDP;
            b[ct].q[0] = *reinterpret_cast<const u32x4*>(wl + k0 + half * 8);
            b[ct].q[1] = *reinterpret_cast<const u32x4*>(wl + k0 + 16 + half * 8);
        }
#pragma unroll
        for (int ct = 0; ct < 4; ++ct) {
            acc0[ct] = __builtin_amdgcn_wmma_f32_16x16x32_bf16(false, a0.v, false, b[ct].v,
                                                               (short)0, acc0[ct], false, false);
            acc1[ct] = __builtin_amdgcn_wmma_f32_16x16x32_bf16(false, a1.v, false, b[ct].v,
                                                               (short)0, acc1[ct], false, false);
        }
    }

#pragma unroll
    for (int ct = 0; ct < 4; ++ct) {
        int cg = col0 + ct * 16 + l16;
        float bc = bias ? bias[cg] : 0.0f;
#pragma unroll
        for (int v = 0; v < 8; ++v) {
            int r0 = rbase + v + half * 8;           // C/D layout
            int r1 = r0 + 128;
            if (r0 < nrows) Y[(size_t)r0 * M + cg] = acc0[ct][v] + bc;
            if (r1 < nrows) Y[(size_t)r1 * M + cg] = acc1[ct][v] + bc;
        }
    }
}

// ---------------------------------------------------------------------------
// Edge pass 1: one wave per edge. score[e,h] = sum_c att[h,c]*lrelu(xl+xr),
// segment max via int-encoded atomicMax into mbuf[dst,h].
// ---------------------------------------------------------------------------
__global__ __launch_bounds__(256)
void edge_score_max(const float* __restrict__ xl, const float* __restrict__ xr,
                    const int* __restrict__ src, const int* __restrict__ dst,
                    const float* __restrict__ att, float* __restrict__ score,
                    int* __restrict__ mbuf, int E)
{
    int g = blockIdx.x * blockDim.x + threadIdx.x;
    int e = g >> 5, lane = g & 31;
    if (e >= E) return;
    int s = src[e], d = dst[e];
    const float* pl = xl + (size_t)s * HC;
    const float* pr = xr + (size_t)d * HC;
    float h0 = 0.f, h1 = 0.f, h2 = 0.f, h3 = 0.f;
#pragma unroll
    for (int chunk = 0; chunk < 8; ++chunk) {
        int idx = chunk * 32 + lane;                 // idx = h*64 + c, coalesced
        float v = pl[idx] + pr[idx];
        v = v > 0.f ? v : 0.2f * v;                  // leaky_relu(0.2)
        float p = v * att[idx];
        if      (chunk < 2) h0 += p;
        else if (chunk < 4) h1 += p;
        else if (chunk < 6) h2 += p;
        else                h3 += p;
    }
#pragma unroll
    for (int o = 16; o; o >>= 1) {
        h0 += __shfl_xor(h0, o, 32);
        h1 += __shfl_xor(h1, o, 32);
        h2 += __shfl_xor(h2, o, 32);
        h3 += __shfl_xor(h3, o, 32);
    }
    if (lane == 0) {
        score[e * 4 + 0] = h0; atomicMax(&mbuf[d * 4 + 0], encf(h0));
        score[e * 4 + 1] = h1; atomicMax(&mbuf[d * 4 + 1], encf(h1));
        score[e * 4 + 2] = h2; atomicMax(&mbuf[d * 4 + 2], encf(h2));
        score[e * 4 + 3] = h3; atomicMax(&mbuf[d * 4 + 3], encf(h3));
    }
}

// ---------------------------------------------------------------------------
// Edge pass 2: w = exp(score - max); atomicAdd sums; scatter xl[src]*w into
// agg[dst] (normalization by the per-dst sum is deferred to finalize).
// ---------------------------------------------------------------------------
__global__ __launch_bounds__(256)
void edge_scatter(const float* __restrict__ xl, const int* __restrict__ src,
                  const int* __restrict__ dst, const float* __restrict__ score,
                  const int* __restrict__ mbuf, float* __restrict__ sbuf,
                  float* __restrict__ agg, int E)
{
    int g = blockIdx.x * blockDim.x + threadIdx.x;
    int e = g >> 5, lane = g & 31;
    if (e >= E) return;
    int s = src[e], d = dst[e];
    float w = 0.f;
    if (lane < 4) {
        float m = decf(mbuf[d * 4 + lane]);
        w = __expf(score[e * 4 + lane] - m);
        atomicAdd(&sbuf[d * 4 + lane], w);
    }
    const float* pl = xl + (size_t)s * HC;
    float*       pd = agg + (size_t)d * HC;
#pragma unroll
    for (int chunk = 0; chunk < 8; ++chunk) {
        int idx = chunk * 32 + lane;
        float wv = __shfl(w, chunk >> 1, 32);        // head = chunk/2, lanes 0..3 hold w
        atomicAdd(&pd[idx], pl[idx] * wv);
    }
}

__global__ void fill_u32(unsigned int* __restrict__ p, unsigned int v, int n)
{
    int i = blockIdx.x * blockDim.x + threadIdx.x;
    if (i < n) p[i] = v;
}

// acc[n,256] += agg[n,256] / (sum[n,h]+eps) + bias[256]   (concat layers)
__global__ void finalize_concat(const float* __restrict__ agg, const float* __restrict__ sbuf,
                                const float* __restrict__ bias, float* __restrict__ acc, int N)
{
    int i = blockIdx.x * blockDim.x + threadIdx.x;
    if (i >= N * HC) return;
    int node = i >> 8, idx = i & 255, h = idx >> 6;
    acc[i] += agg[i] / (sbuf[node * NH + h] + 1e-16f) + bias[idx];
}

// acc[n,64] += mean_h(agg[n,h,64]/(sum[n,h]+eps)) + bias[64]   (layer 3)
__global__ void finalize_mean(const float* __restrict__ agg, const float* __restrict__ sbuf,
                              const float* __restrict__ bias, float* __restrict__ acc, int N)
{
    int i = blockIdx.x * blockDim.x + threadIdx.x;
    if (i >= N * CH) return;
    int node = i >> 6, c = i & 63;
    float s = 0.f;
#pragma unroll
    for (int h = 0; h < NH; ++h)
        s += agg[(size_t)node * HC + h * CH + c] / (sbuf[node * NH + h] + 1e-16f);
    acc[i] += 0.25f * s + bias[c];
}

// out = elu(acc + tile(selfloop))  (in-place safe)
__global__ void selfloop_elu(const float* __restrict__ acc, const float* __restrict__ sl,
                             float* __restrict__ out, int n, int width)
{
    int i = blockIdx.x * blockDim.x + threadIdx.x;
    if (i >= n) return;
    int node = i / width;
    int c = (i % width) & (CH - 1);
    float x = acc[i] + sl[(size_t)node * CH + c];
    out[i] = x > 0.f ? x : (__expf(x) - 1.0f);
}

// ---------------------------------------------------------------------------
// Integrator: per-node view attention + 64->128 relu MLP -> 5 classes.
// One block per node; W1/W2 staged in LDS.
// ---------------------------------------------------------------------------
__global__ __launch_bounds__(128)
void integrator(const float* __restrict__ xm, const float* __restrict__ xi,
                const float* __restrict__ w_att, const float* __restrict__ b_att,
                const float* __restrict__ W1, const float* __restrict__ b1,
                const float* __restrict__ W2, const float* __restrict__ b2,
                float* __restrict__ out, int N)
{
    __shared__ float sW1[CH * 128];
    __shared__ float sW2[128 * 5];
    __shared__ float fused[CH];
    __shared__ float hbuf[128];
    __shared__ float red[CH];
    __shared__ float a2[2];

    const int node = blockIdx.x;
    const int t = threadIdx.x;
    for (int i = t; i < CH * 128; i += 128) sW1[i] = W1[i];
    for (int i = t; i < 128 * 5; i += 128)  sW2[i] = W2[i];

    const float* pm = xm + (size_t)node * CH;
    const float* pi = xi + (size_t)node * CH;

    if (t < CH) red[t] = pm[t] * w_att[t];
    __syncthreads();
    for (int s = 32; s; s >>= 1) { if (t < s) red[t] += red[t + s]; __syncthreads(); }
    if (t == 0) a2[0] = red[0];
    __syncthreads();
    if (t < CH) red[t] = pi[t] * w_att[t];
    __syncthreads();
    for (int s = 32; s; s >>= 1) { if (t < s) red[t] += red[t + s]; __syncthreads(); }
    if (t == 0) {
        float am = a2[0] + b_att[0], ai = red[0] + b_att[0];
        float mx = fmaxf(am, ai);
        float em = __expf(am - mx), ei = __expf(ai - mx);
        float inv = 1.0f / (em + ei);
        a2[0] = em * inv; a2[1] = ei * inv;
    }
    __syncthreads();
    if (t < CH) fused[t] = a2[0] * pm[t] + a2[1] * pi[t];
    __syncthreads();

    float h = b1[t];
#pragma unroll 8
    for (int c = 0; c < CH; ++c) h += fused[c] * sW1[c * 128 + t];
    hbuf[t] = fmaxf(h, 0.f);
    __syncthreads();

    if (t < 5) {
        float o = b2[t];
        for (int j = 0; j < 128; ++j) o += hbuf[j] * sW2[j * 5 + t];
        out[(size_t)node * 5 + t] = o;
    }
}

// ---------------------------------------------------------------------------
static inline int divUp(int a, int b) { return (a + b - 1) / b; }

extern "C" void kernel_launch(void* const* d_in, const int* in_sizes, int n_in,
                              void* d_out, int out_size, void* d_ws, size_t ws_size,
                              hipStream_t stream)
{
    (void)n_in; (void)out_size; (void)ws_size;
    // Flattening assumption (jax pytree, dict keys sorted at every level):
    //  0:x_mrna 1:x_mirna
    //  params: conv1@2, conv2@26, conv3@50 (rels ii,im,mi,mm; leaves Wl,Wr,att,bias,bl,br)
    //          integ@74 (W1,W2,b1,b2,b_att,w_att)  sl1@80 sl2@84 sl3@88 (i.W,i.b,m.W,m.b)
    //  92:edge_mm 93:edge_mi 94:edge_im 95:edge_ii
    const int N = in_sizes[0] / KD;                       // 20000
    auto F = [&](int i) { return (const float*)d_in[i]; };

    char* p = (char*)d_ws;
    auto alloc = [&](size_t bytes) {
        void* r = (void*)p;
        p += (bytes + 255) & ~(size_t)255;
        return r;
    };
    const size_t NF = (size_t)N * HC;
    float* pm[2];  float* pi[2];
    pm[0] = (float*)alloc(NF * 4);  pi[0] = (float*)alloc(NF * 4);
    pm[1] = (float*)alloc(NF * 4);  pi[1] = (float*)alloc(NF * 4);
    float* xlb    = (float*)alloc(NF * 4);
    float* xrb    = (float*)alloc(NF * 4);
    float* agg    = (float*)alloc(NF * 4);
    float* slb    = (float*)alloc((size_t)N * CH * 4);
    float* sbuf   = (float*)alloc((size_t)N * NH * 4);
    int*   mbuf   = (int*)  alloc((size_t)N * NH * 4);
    unsigned short* xbf[2];
    xbf[0] = (unsigned short*)alloc(NF * 2);
    xbf[1] = (unsigned short*)alloc(NF * 2);
    unsigned short* wbf = (unsigned short*)alloc((size_t)KD * HC * 2);
    float* scoreb = (float*)alloc((size_t)(in_sizes[92] / 2) * 4 * 4);

    // sorted relation order ii,im,mi,mm ; type 0=m, 1=i
    const int relS[4] = {1, 1, 0, 0};
    const int relD[4] = {1, 0, 1, 0};
    const int relE[4] = {95, 94, 93, 92};

    const float* cur[2] = { F(0), F(1) };

    for (int layer = 0; layer < 3; ++layer) {
        const int  cb     = 2 + layer * 24;
        const int  sb     = 80 + layer * 4;
        const bool concat = (layer < 2);
        const int  outW   = concat ? HC : CH;
        float* acc[2] = { pm[layer & 1], pi[layer & 1] };

        // activations -> bf16 once per layer (reused by 5 GEMMs each)
        cvt_f32_bf16<<<divUp((int)NF / 8, 256), 256, 0, stream>>>(cur[0], xbf[0], (int)NF);
        cvt_f32_bf16<<<divUp((int)NF / 8, 256), 256, 0, stream>>>(cur[1], xbf[1], (int)NF);

        {   // zero HeteroConv accumulators
            int n = N * outW;
            fill_u32<<<divUp(n, 256), 256, 0, stream>>>((unsigned*)acc[0], 0u, n);
            fill_u32<<<divUp(n, 256), 256, 0, stream>>>((unsigned*)acc[1], 0u, n);
        }

        const int nW = KD * HC;                           // 65536 weight elems
        for (int r = 0; r < 4; ++r) {
            const float* Wl   = F(cb + r * 6 + 0);
            const float* Wr   = F(cb + r * 6 + 1);
            const float* att  = F(cb + r * 6 + 2);
            const float* bias = F(cb + r * 6 + 3);
            const float* bl   = F(cb + r * 6 + 4);
            const float* br   = F(cb + r * 6 + 5);
            const int* ep = (const int*)d_in[relE[r]];
            const int  E  = in_sizes[relE[r]] / 2;
            const int* srcI = ep;
            const int* dstI = ep + E;

            dim3 gg(HC / 64, divUp(N, 256));
            cvt_f32_bf16<<<divUp(nW / 8, 256), 256, 0, stream>>>(Wl, wbf, nW);
            gemm_bf16_wmma<<<gg, 256, 0, stream>>>(xbf[relS[r]], wbf, bl, xlb, N, HC);
            cvt_f32_bf16<<<divUp(nW / 8, 256), 256, 0, stream>>>(Wr, wbf, nW);
            gemm_bf16_wmma<<<gg, 256, 0, stream>>>(xbf[relD[r]], wbf, br, xrb, N, HC);

            fill_u32<<<divUp(N * NH, 256), 256, 0, stream>>>((unsigned*)mbuf, 0x80000000u, N * NH);
            fill_u32<<<divUp(N * NH, 256), 256, 0, stream>>>((unsigned*)sbuf, 0u, N * NH);
            fill_u32<<<divUp((int)NF, 256), 256, 0, stream>>>((unsigned*)agg, 0u, (int)NF);

            int eb = divUp(E * 32, 256);
            edge_score_max<<<eb, 256, 0, stream>>>(xlb, xrb, srcI, dstI, att, scoreb, mbuf, E);
            edge_scatter<<<eb, 256, 0, stream>>>(xlb, srcI, dstI, scoreb, mbuf, sbuf, agg, E);

            if (concat)
                finalize_concat<<<divUp(N * HC, 256), 256, 0, stream>>>(agg, sbuf, bias, acc[relD[r]], N);
            else
                finalize_mean<<<divUp(N * CH, 256), 256, 0, stream>>>(agg, sbuf, bias, acc[relD[r]], N);
        }

        for (int t = 0; t < 2; ++t) {                     // self-loop + ELU
            const float* Wsl = F(sb + (t == 1 ? 0 : 2));
            const float* bsl = F(sb + (t == 1 ? 1 : 3));
            const int nWs = KD * CH;
            cvt_f32_bf16<<<divUp(nWs / 8, 256), 256, 0, stream>>>(Wsl, wbf, nWs);
            dim3 gs(CH / 64, divUp(N, 256));
            gemm_bf16_wmma<<<gs, 256, 0, stream>>>(xbf[t], wbf, bsl, slb, N, CH);
            int n = N * outW;
            selfloop_elu<<<divUp(n, 256), 256, 0, stream>>>(acc[t], slb, acc[t], n, outW);
        }
        cur[0] = acc[0];
        cur[1] = acc[1];
    }

    // integ leaves: W1=74 W2=75 b1=76 b2=77 b_att=78 w_att=79
    integrator<<<N, 128, 0, stream>>>(cur[0], cur[1], F(79), F(78),
                                      F(74), F(76), F(75), F(77),
                                      (float*)d_out, N);
}
